// Attention_29454885715971
// MI455X (gfx1250) — compile-verified, matrix operand order checked by reading, not run on hardware
//
#include <hip/hip_runtime.h>

// ---------------------------------------------------------------------------
// Problem constants (from the reference)
// ---------------------------------------------------------------------------
#define BATCH 2
#define SEQ   2048
#define DMODEL 2048
#define NHEAD 16
#define HDIM  128
#define HHALF 64
#define MROWS (BATCH * SEQ)          // 4096
#define LOG2E 1.4426950408889634f

typedef __attribute__((ext_vector_type(16))) __bf16 v16bf;
typedef __attribute__((ext_vector_type(8)))  float  v8f;
typedef __attribute__((ext_vector_type(4)))  unsigned int u32x4;
typedef int gvec4i __attribute__((vector_size(16)));   // matches builtin param type

#if defined(__has_builtin)
# if __has_builtin(__builtin_amdgcn_global_load_async_to_lds_b128)
#  define HAVE_ASYNC_COPY 1
# endif
#endif
#ifndef HAVE_ASYNC_COPY
# define HAVE_ASYNC_COPY 0
#endif

union FragB16 {
    v16bf v;
    u32x4 q[2];
    unsigned short s[16];
};

__device__ inline unsigned short f2bf(float f) {
    unsigned int u = __builtin_bit_cast(unsigned int, f);
    unsigned int r = u + 0x7FFFu + ((u >> 16) & 1u);   // round-to-nearest-even
    return (unsigned short)(r >> 16);
}
__device__ inline float bf2f(unsigned short h) {
    unsigned int u = ((unsigned int)h) << 16;
    return __builtin_bit_cast(float, u);
}

__device__ inline v8f wmma_bf16(const FragB16& a, const FragB16& b, v8f c) {
    // D = A(16x32 bf16) * B(32x16 bf16) + C(16x16 f32)
    return __builtin_amdgcn_wmma_f32_16x16x32_bf16(
        false, a.v, false, b.v, (short)0, c, false, false);
}

#if HAVE_ASYNC_COPY
// 16B global -> LDS async copy (ASYNCcnt-tracked, bypasses VGPRs)
__device__ inline void async_copy_b128(const unsigned short* g, unsigned short* l) {
    __builtin_amdgcn_global_load_async_to_lds_b128(
        (__attribute__((address_space(1))) gvec4i*)(unsigned long long)(uintptr_t)g,
        (__attribute__((address_space(3))) gvec4i*)(unsigned int)(uintptr_t)l,
        0, 0);
}
#endif

template <int N>
__device__ inline void async_wait() {
#if HAVE_ASYNC_COPY
    asm volatile("s_wait_asynccnt %0" :: "i"(N) : "memory");
#endif
}

// ---------------------------------------------------------------------------
// Kernel 1: fp32 -> bf16 conversion (grid-stride)
// ---------------------------------------------------------------------------
__global__ __launch_bounds__(256) void cvt_f32_bf16(
    const float* __restrict__ in, unsigned short* __restrict__ out, int n)
{
    for (int i = blockIdx.x * blockDim.x + threadIdx.x; i < n;
         i += gridDim.x * blockDim.x)
        out[i] = f2bf(in[i]);
}

// ---------------------------------------------------------------------------
// Kernel 2: GEMM  C[4096,2048] = A[4096,2048] * W[2048,2048]^T (bf16, f32 acc)
// Block tile 128x128, K-step 32, 256 threads = 8 waves, wave tile 32x64.
// LDS double-buffered, tiles streamed in with async copies overlapping WMMA.
// mode 0: store f32 row-major into Cf;  mode 1: store bf16 [b,h,t,d] into Cb
// ---------------------------------------------------------------------------
__global__ __launch_bounds__(256) void gemm_bf16_nt(
    const unsigned short* __restrict__ A,
    const unsigned short* __restrict__ W,
    float* __restrict__ Cf,
    unsigned short* __restrict__ Cb,
    int mode)
{
    __shared__ __align__(16) unsigned short As[2][128 * 32];
    __shared__ __align__(16) unsigned short Ws[2][128 * 32];

    const int tid  = threadIdx.x;
    const int lane = tid & 31;
    const int wid  = tid >> 5;
    const int half = lane >> 4;
    const int lm   = lane & 15;
    const int m0   = blockIdx.y * 128;
    const int n0   = blockIdx.x * 128;
    const int wm   = (wid & 3) * 32;   // wave M offset inside block tile
    const int wn   = (wid >> 2) * 64;  // wave N offset inside block tile

    // two 16B staging slots per thread per matrix (512 slots of 8 bf16 each)
    const int i0 = tid, i1 = tid + 256;
    const unsigned short* Ag0 = A + (size_t)(m0 + (i0 >> 2)) * DMODEL + (i0 & 3) * 8;
    const unsigned short* Ag1 = A + (size_t)(m0 + (i1 >> 2)) * DMODEL + (i1 & 3) * 8;
    const unsigned short* Wg0 = W + (size_t)(n0 + (i0 >> 2)) * DMODEL + (i0 & 3) * 8;
    const unsigned short* Wg1 = W + (size_t)(n0 + (i1 >> 2)) * DMODEL + (i1 & 3) * 8;

    auto stage = [&](int buf, int kt) {
        const int ko = kt * 32;
#if HAVE_ASYNC_COPY
        async_copy_b128(Ag0 + ko, &As[buf][i0 * 8]);
        async_copy_b128(Ag1 + ko, &As[buf][i1 * 8]);
        async_copy_b128(Wg0 + ko, &Ws[buf][i0 * 8]);
        async_copy_b128(Wg1 + ko, &Ws[buf][i1 * 8]);
#else
        const u32x4 a0 = *(const u32x4*)(Ag0 + ko);
        const u32x4 a1 = *(const u32x4*)(Ag1 + ko);
        const u32x4 w0 = *(const u32x4*)(Wg0 + ko);
        const u32x4 w1 = *(const u32x4*)(Wg1 + ko);
        ((u32x4*)As[buf])[i0] = a0;
        ((u32x4*)As[buf])[i1] = a1;
        ((u32x4*)Ws[buf])[i0] = w0;
        ((u32x4*)Ws[buf])[i1] = w1;
#endif
    };

    v8f acc[2][4];
    for (int i = 0; i < 2; ++i)
        for (int j = 0; j < 4; ++j)
            acc[i][j] = (v8f){0.f,0.f,0.f,0.f,0.f,0.f,0.f,0.f};

    const int nkt = DMODEL / 32;   // 64
    stage(0, 0);                   // prologue

    for (int kt = 0; kt < nkt; ++kt) {
        const int cur = kt & 1;
        __syncthreads();                       // buf cur^1 free to overwrite
        if (kt + 1 < nkt) {
            stage(cur ^ 1, kt + 1);            // stream next tile behind compute
            __builtin_prefetch(Ag0 + (kt + 2) * 32, 0, 0);
            async_wait<4>();                   // tile kt landed (in-order)
        } else {
            async_wait<0>();
        }
        __syncthreads();                       // tile kt visible to all waves

        FragB16 a[2], b[4];
        for (int i = 0; i < 2; ++i) {
            const unsigned short* row = As[cur] + (wm + i * 16 + lm) * 32;
            a[i].q[0] = *(const u32x4*)(row + half * 8);
            a[i].q[1] = *(const u32x4*)(row + 16 + half * 8);
        }
        for (int j = 0; j < 4; ++j) {
            const unsigned short* row = Ws[cur] + (wn + j * 16 + lm) * 32;
            b[j].q[0] = *(const u32x4*)(row + half * 8);
            b[j].q[1] = *(const u32x4*)(row + 16 + half * 8);
        }
        for (int i = 0; i < 2; ++i)
            for (int j = 0; j < 4; ++j)
                acc[i][j] = wmma_bf16(a[i], b[j], acc[i][j]);
    }

    // epilogue
    for (int i = 0; i < 2; ++i)
        for (int j = 0; j < 4; ++j)
            for (int e = 0; e < 8; ++e) {
                const int row = m0 + wm + i * 16 + e + 8 * half;
                const int col = n0 + wn + j * 16 + lm;
                const float v = acc[i][j][e];
                if (mode == 0) {
                    Cf[(size_t)row * DMODEL + col] = v;
                } else {
                    const int bb = row >> 11, t = row & (SEQ - 1);
                    const int h  = col >> 7,  d = col & (HDIM - 1);
                    Cb[(((size_t)(bb * NHEAD + h)) * SEQ + t) * HDIM + d] = f2bf(v);
                }
            }
}

// ---------------------------------------------------------------------------
// Kernel 3: RoPE in-place on bf16 [b,h,t,d] tensor (rotate-half formulation)
// ---------------------------------------------------------------------------
__global__ __launch_bounds__(256) void rope_bf16(
    unsigned short* __restrict__ qk,
    const float* __restrict__ cosb, const float* __restrict__ sinb, int n)
{
    for (int i = blockIdx.x * blockDim.x + threadIdx.x; i < n;
         i += gridDim.x * blockDim.x) {
        const int j  = i & (HHALF - 1);
        const int t  = (i >> 6) & (SEQ - 1);
        const int bh = i >> 17;
        const size_t base = ((size_t)bh * SEQ + t) * HDIM;
        const float x1 = bf2f(qk[base + j]);
        const float x2 = bf2f(qk[base + HHALF + j]);
        const float c  = cosb[t * HHALF + j];
        const float s  = sinb[t * HHALF + j];
        qk[base + j]         = f2bf(x1 * c - x2 * s);
        qk[base + HHALF + j] = f2bf(x1 * s + x2 * c);
    }
}

// ---------------------------------------------------------------------------
// Kernel 4: causal flash attention, bf16 QKV -> bf16 O ([b,t,h*d] layout)
// grid = (T/64, B*H), 128 threads = 4 waves, each wave owns 16 query rows.
// ---------------------------------------------------------------------------
__global__ __launch_bounds__(128) void flash_attn(
    const unsigned short* __restrict__ Qg,
    const unsigned short* __restrict__ Kg,
    const unsigned short* __restrict__ Vg,
    unsigned short* __restrict__ Og)
{
    __shared__ __align__(16) unsigned short Ks[32 * 128];     // [k][d]
    __shared__ __align__(16) unsigned short Vt[128 * 32];     // [d][k] transposed
    __shared__ __align__(16) unsigned short Ps[4][16 * 32];   // per-wave P tile

    const int tid  = threadIdx.x;
    const int lane = tid & 31;
    const int wid  = tid >> 5;
    const int half = lane >> 4;
    const int lm   = lane & 15;
    const int bh   = blockIdx.y;                 // b*H + h
    const int q0   = blockIdx.x * 64 + wid * 16; // wave's first query row
    const size_t base = (size_t)bh * SEQ * HDIM;

    // Q fragments: 16 rows x 128 cols -> 4 A-frags (K-chunks of 32)
    FragB16 qa[4];
    for (int f = 0; f < 4; ++f) {
        const unsigned short* r = Qg + base + (size_t)(q0 + lm) * HDIM + f * 32;
        qa[f].q[0] = *(const u32x4*)(r + half * 8);
        qa[f].q[1] = *(const u32x4*)(r + 16 + half * 8);
    }

    v8f o[8];
    float mx[8], ls[8];
    for (int f = 0; f < 8; ++f) o[f] = (v8f){0.f,0.f,0.f,0.f,0.f,0.f,0.f,0.f};
    for (int i = 0; i < 8; ++i) { mx[i] = -3.0e38f; ls[i] = 0.f; }

    const float scale = 0.08838834764831845f;    // 1/sqrt(128)
    const int ntiles = (blockIdx.x + 1) * 2;     // 32-key tiles (causal bound)

    for (int kt = 0; kt < ntiles; ++kt) {
        const int k0 = kt * 32;
        const unsigned short* kb = Kg + base + (size_t)k0 * HDIM;
        __syncthreads();
        // stage K tile [32][128]: 4 x 16B per thread
#if HAVE_ASYNC_COPY
        {
            const int s0 = tid, s1 = tid + 128, s2 = tid + 256, s3 = tid + 384;
            async_copy_b128(kb + (s0 >> 4) * HDIM + (s0 & 15) * 8, Ks + s0 * 8);
            async_copy_b128(kb + (s1 >> 4) * HDIM + (s1 & 15) * 8, Ks + s1 * 8);
            async_copy_b128(kb + (s2 >> 4) * HDIM + (s2 & 15) * 8, Ks + s2 * 8);
            async_copy_b128(kb + (s3 >> 4) * HDIM + (s3 & 15) * 8, Ks + s3 * 8);
        }
#else
        {
            const int s0 = tid, s1 = tid + 128, s2 = tid + 256, s3 = tid + 384;
            const u32x4 k0v = *(const u32x4*)(kb + (s0 >> 4) * HDIM + (s0 & 15) * 8);
            const u32x4 k1v = *(const u32x4*)(kb + (s1 >> 4) * HDIM + (s1 & 15) * 8);
            const u32x4 k2v = *(const u32x4*)(kb + (s2 >> 4) * HDIM + (s2 & 15) * 8);
            const u32x4 k3v = *(const u32x4*)(kb + (s3 >> 4) * HDIM + (s3 & 15) * 8);
            ((u32x4*)Ks)[s0] = k0v;
            ((u32x4*)Ks)[s1] = k1v;
            ((u32x4*)Ks)[s2] = k2v;
            ((u32x4*)Ks)[s3] = k3v;
        }
#endif
        // stage V tile transposed [128][32] (feeds contiguous B-frag runs)
        for (int i = tid; i < 4096; i += 128) {
            const int r = i >> 7, d = i & 127;
            Vt[d * 32 + r] = Vg[base + (size_t)(k0 + r) * HDIM + d];
        }
        async_wait<0>();
        __syncthreads();

        if (k0 <= q0 + 15) {     // tile has at least one unmasked key (wave-uniform)
            // S = Q * K^T : two 16x16 f32 tiles (keys 0-15 and 16-31)
            v8f s0 = (v8f){0.f,0.f,0.f,0.f,0.f,0.f,0.f,0.f};
            v8f s1 = s0;
            for (int f = 0; f < 4; ++f) {
                FragB16 kfr;
                const unsigned short* r0 = Ks + (size_t)lm * HDIM + f * 32;
                kfr.q[0] = *(const u32x4*)(r0 + half * 8);
                kfr.q[1] = *(const u32x4*)(r0 + 16 + half * 8);
                s0 = wmma_bf16(qa[f], kfr, s0);
                const unsigned short* r1 = Ks + (size_t)(16 + lm) * HDIM + f * 32;
                kfr.q[0] = *(const u32x4*)(r1 + half * 8);
                kfr.q[1] = *(const u32x4*)(r1 + 16 + half * 8);
                s1 = wmma_bf16(qa[f], kfr, s1);
            }

            // online softmax (rows live in VGPR index + lane-half; N across lanes)
            float p0[8], p1[8], alpha[8];
            const int kk0 = k0 + lm, kk1 = k0 + 16 + lm;
            for (int i = 0; i < 8; ++i) {
                const int qrow = q0 + i + 8 * half;
                float v0 = s0[i] * scale; if (kk0 > qrow) v0 = -3.0e38f;
                float v1 = s1[i] * scale; if (kk1 > qrow) v1 = -3.0e38f;
                float rm = fmaxf(v0, v1);
                rm = fmaxf(rm, __shfl_xor(rm, 1, 32));
                rm = fmaxf(rm, __shfl_xor(rm, 2, 32));
                rm = fmaxf(rm, __shfl_xor(rm, 4, 32));
                rm = fmaxf(rm, __shfl_xor(rm, 8, 32));
                const float mnew = fmaxf(mx[i], rm);
                const float a = __builtin_amdgcn_exp2f((mx[i] - mnew) * LOG2E);
                const float e0 = (kk0 > qrow) ? 0.f
                    : __builtin_amdgcn_exp2f((v0 - mnew) * LOG2E);
                const float e1 = (kk1 > qrow) ? 0.f
                    : __builtin_amdgcn_exp2f((v1 - mnew) * LOG2E);
                float rs = e0 + e1;
                rs += __shfl_xor(rs, 1, 32);
                rs += __shfl_xor(rs, 2, 32);
                rs += __shfl_xor(rs, 4, 32);
                rs += __shfl_xor(rs, 8, 32);
                ls[i] = ls[i] * a + rs;
                mx[i] = mnew;
                alpha[i] = a; p0[i] = e0; p1[i] = e1;
            }
            for (int f = 0; f < 8; ++f)
                for (int i = 0; i < 8; ++i)
                    o[f][i] *= alpha[i];

            // transpose P (C layout -> A layout) through per-wave LDS buffer
            unsigned short* Pw = Ps[wid];
            for (int i = 0; i < 8; ++i) {
                const int row = i + 8 * half;
                Pw[row * 32 + lm]      = f2bf(p0[i]);
                Pw[row * 32 + 16 + lm] = f2bf(p1[i]);
            }
            asm volatile("s_wait_dscnt 0x0" ::: "memory");  // wave-local LDS RAW
            FragB16 pa;
            const unsigned short* pr = Pw + lm * 32;
            pa.q[0] = *(const u32x4*)(pr + half * 8);
            pa.q[1] = *(const u32x4*)(pr + 16 + half * 8);

            // O += P * V  (V transposed tile gives contiguous B-frag runs)
            for (int f = 0; f < 8; ++f) {
                FragB16 vb;
                const unsigned short* vr = Vt + (f * 16 + lm) * 32;
                vb.q[0] = *(const u32x4*)(vr + half * 8);
                vb.q[1] = *(const u32x4*)(vr + 16 + half * 8);
                o[f] = wmma_bf16(pa, vb, o[f]);
            }
        }
    }

    // epilogue: normalize and store O as [b, t, h*128+d] bf16
    const int bb = bh >> 4, h = bh & 15;
    for (int i = 0; i < 8; ++i) {
        const float inv = ls[i] > 0.f ? 1.f / ls[i] : 0.f;
        const int t = q0 + i + 8 * half;
        for (int f = 0; f < 8; ++f) {
            const int col = h * HDIM + f * 16 + lm;
            Og[((size_t)(bb * SEQ + t)) * DMODEL + col] = f2bf(o[f][i] * inv);
        }
    }
}

// ---------------------------------------------------------------------------
// Launch
// ---------------------------------------------------------------------------
extern "C" void kernel_launch(void* const* d_in, const int* in_sizes, int n_in,
                              void* d_out, int out_size, void* d_ws, size_t ws_size,
                              hipStream_t stream) {
    const float* x    = (const float*)d_in[0];
    const float* wq   = (const float*)d_in[1];
    const float* wk   = (const float*)d_in[2];
    const float* wv   = (const float*)d_in[3];
    const float* wo   = (const float*)d_in[4];
    const float* cosb = (const float*)d_in[5];
    const float* sinb = (const float*)d_in[6];
    float* out = (float*)d_out;

    unsigned short* ws = (unsigned short*)d_ws;
    size_t off = 0;
    unsigned short* xb  = ws + off; off += (size_t)MROWS * DMODEL;
    unsigned short* wqb = ws + off; off += (size_t)DMODEL * DMODEL;
    unsigned short* wkb = ws + off; off += (size_t)DMODEL * DMODEL;
    unsigned short* wvb = ws + off; off += (size_t)DMODEL * DMODEL;
    unsigned short* wob = ws + off; off += (size_t)DMODEL * DMODEL;
    unsigned short* Qb  = ws + off; off += (size_t)BATCH * NHEAD * SEQ * HDIM;
    unsigned short* Kb  = ws + off; off += (size_t)BATCH * NHEAD * SEQ * HDIM;
    unsigned short* Vb  = ws + off; off += (size_t)BATCH * NHEAD * SEQ * HDIM;
    unsigned short* Ob  = ws + off; off += (size_t)MROWS * DMODEL;

    // 1) fp32 -> bf16
    cvt_f32_bf16<<<4096, 256, 0, stream>>>(x,  xb,  MROWS * DMODEL);
    cvt_f32_bf16<<<2048, 256, 0, stream>>>(wq, wqb, DMODEL * DMODEL);
    cvt_f32_bf16<<<2048, 256, 0, stream>>>(wk, wkb, DMODEL * DMODEL);
    cvt_f32_bf16<<<2048, 256, 0, stream>>>(wv, wvb, DMODEL * DMODEL);
    cvt_f32_bf16<<<2048, 256, 0, stream>>>(wo, wob, DMODEL * DMODEL);

    // 2) QKV projections (store into [b,h,t,d] bf16)
    dim3 gg(DMODEL / 128, MROWS / 128);  // (16, 32)
    gemm_bf16_nt<<<gg, 256, 0, stream>>>(xb, wqb, nullptr, Qb, 1);
    gemm_bf16_nt<<<gg, 256, 0, stream>>>(xb, wkb, nullptr, Kb, 1);
    gemm_bf16_nt<<<gg, 256, 0, stream>>>(xb, wvb, nullptr, Vb, 1);

    // 3) RoPE on Q and K
    const int nrope = BATCH * NHEAD * SEQ * HHALF;
    rope_bf16<<<4096, 256, 0, stream>>>(Qb, cosb, sinb, nrope);
    rope_bf16<<<4096, 256, 0, stream>>>(Kb, cosb, sinb, nrope);

    // 4) causal flash attention -> Ob [b, t, h*d]
    dim3 fg(SEQ / 64, BATCH * NHEAD);    // (32, 32)
    flash_attn<<<fg, 128, 0, stream>>>(Qb, Kb, Vb, Ob);

    // 5) output projection (f32 result straight to d_out)
    gemm_bf16_nt<<<gg, 256, 0, stream>>>(Ob, wob, out, nullptr, 0);
}
